// AttentionAggregator_20641612825109
// MI455X (gfx1250) — compile-verified, matrix operand order checked by reading, not run on hardware
//
#include <hip/hip_runtime.h>
#include <hip/hip_bf16.h>

// ---------------------------------------------------------------------------
// AttentionAggregator for MI455X (gfx1250, wave32, WMMA).
// lin = x @ W_lin via v_wmma_f32_16x16x32_bf16 (fp32 accumulate),
// attention logits via a second WMMA GEMM, softmax in-LDS, fused scale+relu
// output with non-temporal float4 stores. One workgroup (8 waves) per batch.
// LDS rows padded 16B to kill ds_load_b128 bank conflicts (64 banks x 4B).
// ---------------------------------------------------------------------------

typedef __bf16 v16bf __attribute__((ext_vector_type(16)));
typedef __bf16 v8bf  __attribute__((ext_vector_type(8)));
typedef __bf16 v4bf  __attribute__((ext_vector_type(4)));
typedef float  v8f   __attribute__((ext_vector_type(8)));
typedef float  v4f   __attribute__((ext_vector_type(4)));

#define BATCH 4096
#define NN    64
#define FF    128
#define HH    8
#define HD    512

// Padded LDS strides (bytes): stride % 256B != 0 so half-wave b128 gathers
// at row*stride hit disjoint 4-bank windows.
#define LIN_STRIDE 1040               // 512 bf16 + 8 pad  (260 dwords % 64 = 4)
#define XBF_STRIDE 272                // 128 bf16 + 8 pad  ( 68 dwords % 64 = 4)
#define LOGW       17                 // logits row = 17 floats (pad from 16)

// LDS layout (dynamic shared)
#define LINBF_OFF 0                                   // 64*1040 = 66560 B
#define XBF_OFF   66560                               // 64*272  = 17408 B
#define LOG_OFF   (66560 + 17408)                     // 64*17*4 =  4352 B
#define ASRC_OFF  (LOG_OFF + 4352)                    //             64 B
#define SMEM_BYTES (ASRC_OFF + 64)

__device__ __forceinline__ v16bf cat16(v8bf a, v8bf b) {
  return __builtin_shufflevector(a, b, 0,1,2,3,4,5,6,7,8,9,10,11,12,13,14,15);
}
// A fragment: lane<16 holds K{0..7} at +0 and K{16..23} at +32B;
// lane>=16 start is +16B (K{8..15}, K{24..31}). Caller folds the +16 into off.
__device__ __forceinline__ v16bf load_a16(const char* lds, int byteOff) {
  v8bf a = *(const v8bf*)(lds + byteOff);
  v8bf b = *(const v8bf*)(lds + byteOff + 32);
  return cat16(a, b);
}
// B fragment: pre-packed in workspace, 16 contiguous halfs per lane.
__device__ __forceinline__ v16bf load_b16(const __bf16* g) {
  v8bf a = *(const v8bf*)g;
  v8bf b = *(const v8bf*)(g + 8);
  return cat16(a, b);
}

// ---------------------------------------------------------------------------
// Kernel 1: pack W_lin (128x512) and W_att[512:] (512x8, zero-padded to 16
// cols) into bf16 B-fragment-linear layout: frag f -> 32 lanes x 16 halfs.
// Lane = N column (lane&15); within-lane K order mirrors the A-matrix layout.
// ---------------------------------------------------------------------------
__global__ void pack_weights_kernel(const float* __restrict__ W_lin,
                                    const float* __restrict__ W_att,
                                    __bf16* __restrict__ wsB,
                                    __bf16* __restrict__ wsA2) {
  int f = blockIdx.x;         // 0..127 : W_lin frags ; 128..143 : W_att frags
  int t = threadIdx.x;        // 256 threads, 2 halfs each (512 halfs / frag)
  #pragma unroll
  for (int s = 0; s < 2; ++s) {
    int q    = t * 2 + s;
    int lane = q >> 4;
    int j    = q & 15;
    int kl   = (j & 7) + ((j >> 3) << 4) + ((lane >= 16) ? 8 : 0);
    int col  = lane & 15;
    if (f < 128) {
      int kt = f >> 5, ntg = f & 31;
      int k = kt * 32 + kl, n = ntg * 16 + col;
      wsB[f * 512 + q] = (__bf16)W_lin[k * HD + n];
    } else {
      int kt = f - 128;
      int k = kt * 32 + kl;
      float v = (col < HH) ? W_att[(HD + k) * HH + col] : 0.0f;
      wsA2[kt * 512 + q] = (__bf16)v;
    }
  }
}

// ---------------------------------------------------------------------------
// Kernel 2: one block per batch element.
// ---------------------------------------------------------------------------
__global__ __launch_bounds__(256)
void attn_main_kernel(const float* __restrict__ x,
                      const float* __restrict__ W_att,
                      const int*   __restrict__ mask,
                      const __bf16* __restrict__ wsB,
                      const __bf16* __restrict__ wsA2,
                      float* __restrict__ out) {
  extern __shared__ char smem[];
  const int b    = blockIdx.x;
  const int t    = threadIdx.x;
  const int wave = t >> 5;
  const int lane = t & 31;
  const int lo   = lane & 15;
  const int hi   = lane >> 4;   // 0 or 1

  // ---- Phase 0: x[b] (64x128 f32) -> LDS bf16; NT loads (read-once) ------
  {
    const v4f* xg = (const v4f*)(x + (size_t)b * (NN * FF));
    char* xb = smem + XBF_OFF;
    #pragma unroll
    for (int i = 0; i < 8; ++i) {
      int fi = i * 256 + t;
      v4f v = __builtin_nontemporal_load(xg + fi);
      v4bf p;
      p[0] = (__bf16)v[0]; p[1] = (__bf16)v[1];
      p[2] = (__bf16)v[2]; p[3] = (__bf16)v[3];
      int e = fi * 4;                       // element index in 64x128
      int row = e >> 7, colh = e & 127;
      *(v4bf*)(xb + row * XBF_STRIDE + colh * 2) = p;
    }
  }
  __syncthreads();

  // ---- Phase 1: lin = x @ W_lin. Wave w owns head w (cols 64w..64w+63). ---
  {
    const char* xls = smem + XBF_OFF;
    __bf16* linbase = (__bf16*)(smem + LINBF_OFF);
    for (int nt = 0; nt < 4; ++nt) {
      const int ntg = wave * 4 + nt;
      v16bf Bf[4];
      #pragma unroll
      for (int kt = 0; kt < 4; ++kt)
        Bf[kt] = load_b16(wsB + (kt * 32 + ntg) * 512 + lane * 16);
      #pragma unroll
      for (int mt = 0; mt < 4; ++mt) {
        v8f acc = {};
        #pragma unroll
        for (int kt = 0; kt < 4; ++kt) {
          v16bf Af = load_a16(xls, (mt * 16 + lo) * XBF_STRIDE + kt * 64 + hi * 16);
          acc = __builtin_amdgcn_wmma_f32_16x16x32_bf16(
              false, Af, false, Bf[kt], (short)0, acc, false, false);
        }
        const int col = wave * 64 + nt * 16 + lo;
        #pragma unroll
        for (int r = 0; r < 8; ++r) {        // C layout: VGPR r -> M=r / r+8
          int m = mt * 16 + r + hi * 8;
          *(__bf16*)((char*)linbase + m * LIN_STRIDE + col * 2) = (__bf16)acc[r];
        }
      }
    }
  }
  __syncthreads();

  // ---- Phase 2: logits = lin @ W_att[512:] (waves 0-3), src bias (wave 4) --
  if (wave < 4) {
    const char* lls = smem + LINBF_OFF;
    float* lg = (float*)(smem + LOG_OFF);
    v8f aacc = {};
    #pragma unroll
    for (int kt = 0; kt < 16; ++kt) {          // K = 512
      v16bf Af = load_a16(lls, (wave * 16 + lo) * LIN_STRIDE + kt * 64 + hi * 16);
      v16bf Bf = load_b16(wsA2 + kt * 512 + lane * 16);
      aacc = __builtin_amdgcn_wmma_f32_16x16x32_bf16(
          false, Af, false, Bf, (short)0, aacc, false, false);
    }
    #pragma unroll
    for (int r = 0; r < 8; ++r) {
      int n = wave * 16 + r + hi * 8;
      lg[n * LOGW + lo] = aacc[r];             // raw logits (pre src/leaky)
    }
  } else if (wave == 4) {
    // att_src[h] = lin[0,:] . W_att[:512, h]
    const __bf16* l0 = (const __bf16*)(smem + LINBF_OFF);   // row 0, no pad
    float s[8] = {0, 0, 0, 0, 0, 0, 0, 0};
    for (int c = lane; c < HD; c += 32) {
      float lv = (float)l0[c];
      const v4f* wr = (const v4f*)(W_att + c * HH);
      v4f w0 = wr[0], w1 = wr[1];
      s[0] += lv * w0[0]; s[1] += lv * w0[1]; s[2] += lv * w0[2]; s[3] += lv * w0[3];
      s[4] += lv * w1[0]; s[5] += lv * w1[1]; s[6] += lv * w1[2]; s[7] += lv * w1[3];
    }
    #pragma unroll
    for (int i = 0; i < 8; ++i)
      for (int m = 16; m >= 1; m >>= 1) s[i] += __shfl_xor(s[i], m, 32);
    if (lane == 0) {
      float* as = (float*)(smem + ASRC_OFF);
      #pragma unroll
      for (int i = 0; i < 8; ++i) as[i] = s[i];
    }
  }
  __syncthreads();

  // ---- Phase 3: per-head softmax over n (wave w -> head w), in place -----
  {
    float* lg = (float*)(smem + LOG_OFF);
    const float sA = ((const float*)(smem + ASRC_OFF))[wave];
    const int n0 = lane, n1 = lane + 32;
    float l0 = lg[n0 * LOGW + wave] + sA;
    float l1 = lg[n1 * LOGW + wave] + sA;
    l0 = (l0 >= 0.f) ? l0 : 0.2f * l0;         // leaky relu
    l1 = (l1 >= 0.f) ? l1 : 0.2f * l1;
    const int* mk = mask + b * NN;
    if (mk[n0] == 0) l0 -= 1e9f;
    if (mk[n1] == 0) l1 -= 1e9f;
    float mx = fmaxf(l0, l1);
    for (int m = 16; m >= 1; m >>= 1) mx = fmaxf(mx, __shfl_xor(mx, m, 32));
    float e0 = __expf(l0 - mx), e1 = __expf(l1 - mx);
    float sm = e0 + e1;
    for (int m = 16; m >= 1; m >>= 1) sm += __shfl_xor(sm, m, 32);
    float inv = __frcp_rn(sm);
    lg[n0 * LOGW + wave] = e0 * inv;           // own column only -> race-free
    lg[n1 * LOGW + wave] = e1 * inv;
  }
  __syncthreads();

  // ---- Phase 4: out = relu(lin * aw); NT float4 stores (write-once) ------
  {
    const char*  linb = smem + LINBF_OFF;
    const float* lg   = (const float*)(smem + LOG_OFF);
    v4f* og = (v4f*)(out + (size_t)b * (NN * HD));
    #pragma unroll 4
    for (int i = 0; i < 32; ++i) {
      int idx = i * 1024 + t * 4;              // linear over 64*512 elements
      int n = idx >> 9, c = idx & 511, h = c >> 6;
      v4bf lv = *(const v4bf*)(linb + n * LIN_STRIDE + c * 2);
      float a = lg[n * LOGW + h];
      v4f o;
      o[0] = fmaxf((float)lv[0] * a, 0.f);
      o[1] = fmaxf((float)lv[1] * a, 0.f);
      o[2] = fmaxf((float)lv[2] * a, 0.f);
      o[3] = fmaxf((float)lv[3] * a, 0.f);
      __builtin_nontemporal_store(o, og + (idx >> 2));
    }
  }
}

extern "C" void kernel_launch(void* const* d_in, const int* in_sizes, int n_in,
                              void* d_out, int out_size, void* d_ws, size_t ws_size,
                              hipStream_t stream) {
  const float* x     = (const float*)d_in[0];
  const float* W_lin = (const float*)d_in[1];
  const float* W_att = (const float*)d_in[2];
  const int*   mask  = (const int*)d_in[3];
  float* out = (float*)d_out;

  __bf16* wsB  = (__bf16*)d_ws;                       // 128 KB packed W_lin
  __bf16* wsA2 = (__bf16*)((char*)d_ws + 131072);     //  16 KB packed W_att2

  pack_weights_kernel<<<144, 256, 0, stream>>>(W_lin, W_att, wsB, wsA2);

  (void)hipFuncSetAttribute((const void*)attn_main_kernel,
                            hipFuncAttributeMaxDynamicSharedMemorySize,
                            SMEM_BYTES);
  attn_main_kernel<<<BATCH, 256, SMEM_BYTES, stream>>>(x, W_att, mask, wsB,
                                                       wsA2, out);
}
